// GMoEExpert_55542517072590
// MI455X (gfx1250) — matrix operand order repaired and checked by reading, not compiled, not run on hardware
//
#include <hip/hip_runtime.h>

typedef __attribute__((ext_vector_type(16))) __bf16 v16bf;
typedef __attribute__((ext_vector_type(8)))  float  v8f;

#define THREADS 256
#define TILE_T  16
#define CHUNK   512      // N columns staged per pass (8 waves * 4 tiles * 16)
#define KSTEP   32       // K per v_wmma_f32_16x16x32_bf16
#define NT      4        // 16-wide n-tiles per wave per chunk

#define E_DIM 8
#define T_DIM 2048
#define D_DIM 1024
#define H_DIM 4096

// LDS strides in bf16 elements, padded so row step is an odd multiple of 16B
// (64 banks x 4B): consecutive rows land on distinct banks, and all
// fragment reads stay 16B-aligned for ds_load_b128.
#define XLDS_STRIDE 1032   // 16 rows  x 1032 -> 33024 B
#define ACT_STRIDE  4104   // 16 rows  x 4104 -> 131328 B
#define SLAB_STRIDE 40     // 512 cols x 40   -> 40960 B (transposed: [col][k])

#define LDS_X_OFF     0
#define LDS_ACT_OFF   33024
#define LDS_SLAB0_OFF 164352
#define LDS_SLAB1_OFF 205312
#define LDS_TOTAL     246272

// Paired f32->bf16 conversion: force a single v_cvt_pk_bf16_f32 (the
// clang builtin is not declared for gfx1250; scalar casts lower to
// half-dead packs + mov_b16/or reassembly, ~5 VALU per pair).
__device__ __forceinline__ unsigned pack_bf16(float a, float b) {
  unsigned r;
  asm("v_cvt_pk_bf16_f32 %0, %1, %2" : "=v"(r) : "v"(a), "v"(b));
  return r;
}

// ---- Stage x tile (fp32 global, row-major) -> bf16 LDS, padded stride ----
__device__ __forceinline__ void stage_x(const float* __restrict__ g,
                                        __bf16* lds, int tid) {
#pragma unroll
  for (int i = 0; i < (TILE_T * D_DIM) / (THREADS * 4); ++i) {  // 16 iters
    const int idx = (tid + i * THREADS) * 4;
    const int r = idx >> 10;       // / D_DIM
    const int c = idx & 1023;
    const float4 f = *(const float4*)(g + (size_t)r * D_DIM + c);
    uint2 u;
    u.x = pack_bf16(f.x, f.y);
    u.y = pack_bf16(f.z, f.w);
    *(uint2*)((unsigned short*)lds + (size_t)r * XLDS_STRIDE + c) = u;
  }
}

// ---- Double-buffered slab pipeline: 32(K) x 512(N) fp32 -> bf16 LDS^T ----
// slab_load issues 16 coalesced global_load_b128 into registers;
// slab_store does the 4x4 register transpose + paired bf16 packs and
// writes ds_store_b64 of 4 consecutive K per column.
struct Staged { float4 r[16]; };

__device__ __forceinline__ void slab_load(Staged& s, const float* __restrict__ g,
                                          int gstride, int tid) {
#pragma unroll
  for (int i = 0; i < 4; ++i) {
    const int q  = tid + i * THREADS;   // 0..1023: 128 col-groups x 8 k-groups
    const int cg = q & 127;
    const int kq = q >> 7;
    const float* gp = g + (size_t)(kq * 4) * gstride + cg * 4;
    s.r[i * 4 + 0] = *(const float4*)(gp);
    s.r[i * 4 + 1] = *(const float4*)(gp + gstride);
    s.r[i * 4 + 2] = *(const float4*)(gp + 2 * (size_t)gstride);
    s.r[i * 4 + 3] = *(const float4*)(gp + 3 * (size_t)gstride);
  }
}

__device__ __forceinline__ void slab_store(const Staged& s, __bf16* slabT,
                                           int tid) {
#pragma unroll
  for (int i = 0; i < 4; ++i) {
    const int q  = tid + i * THREADS;
    const int cg = q & 127;
    const int kq = q >> 7;
    const float4 r0 = s.r[i * 4 + 0];
    const float4 r1 = s.r[i * 4 + 1];
    const float4 r2 = s.r[i * 4 + 2];
    const float4 r3 = s.r[i * 4 + 3];
    unsigned short* dst = (unsigned short*)slabT + kq * 4;
    uint2 u;
    u.x = pack_bf16(r0.x, r1.x); u.y = pack_bf16(r2.x, r3.x);
    *(uint2*)(dst + (size_t)(cg * 4 + 0) * SLAB_STRIDE) = u;
    u.x = pack_bf16(r0.y, r1.y); u.y = pack_bf16(r2.y, r3.y);
    *(uint2*)(dst + (size_t)(cg * 4 + 1) * SLAB_STRIDE) = u;
    u.x = pack_bf16(r0.z, r1.z); u.y = pack_bf16(r2.z, r3.z);
    *(uint2*)(dst + (size_t)(cg * 4 + 2) * SLAB_STRIDE) = u;
    u.x = pack_bf16(r0.w, r1.w); u.y = pack_bf16(r2.w, r3.w);
    *(uint2*)(dst + (size_t)(cg * 4 + 3) * SLAB_STRIDE) = u;
  }
}

// ---- A fragment: 16x32 (MxK) bf16, wave32 (ISA 7.12.2) ----
// Per lane: row m = lane&15, two contiguous K runs of 8 -> 2x ds_load_b128.
__device__ __forceinline__ v16bf load_frag_a(const __bf16* src, int row_stride,
                                             int kbase, int lane) {
  const int m   = lane & 15;
  const int hlf = lane >> 4;
  union { uint4 q[2]; v16bf v; } r;
  const unsigned short* s =
      (const unsigned short*)src + (size_t)m * row_stride + kbase + 8 * hlf;
  r.q[0] = *(const uint4*)(s);
  r.q[1] = *(const uint4*)(s + 16);
  return r.v;
}

// ---- B fragment: 32x16 (KxN) bf16, wave32 ----
// Per lane: column n, K = 8*half..8*half+7 and 16+8*half..16+8*half+7
// in the transposed slab -> 2x ds_load_b128.
__device__ __forceinline__ v16bf load_frag_b(const __bf16* slabT, int col,
                                             int lane) {
  const int n   = lane & 15;
  const int hlf = lane >> 4;
  union { uint4 q[2]; v16bf v; } r;
  const unsigned short* s =
      (const unsigned short*)slabT + (size_t)(col + n) * SLAB_STRIDE + 8 * hlf;
  r.q[0] = *(const uint4*)(s);
  r.q[1] = *(const uint4*)(s + 16);
  return r.v;
}

// ---------------- Fused MoE expert FFN ----------------
__global__ __launch_bounds__(THREADS)
void moe_ffn_fused_kernel(const float* __restrict__ x,
                          const float* __restrict__ w1,
                          const float* __restrict__ b1,
                          const float* __restrict__ w2,
                          const float* __restrict__ b2,
                          float* __restrict__ out) {
  extern __shared__ char smem[];
  __bf16* x_lds   = (__bf16*)(smem + LDS_X_OFF);      // [16][1032]
  __bf16* act_lds = (__bf16*)(smem + LDS_ACT_OFF);    // [16][4104]
  __bf16* slab0   = (__bf16*)(smem + LDS_SLAB0_OFF);  // [512][40] transposed
  __bf16* slab1   = (__bf16*)(smem + LDS_SLAB1_OFF);  // [512][40] transposed

  const int tid  = threadIdx.x;
  const int lane = tid & 31;           // wave32
  const int wave = tid >> 5;
  const int n    = lane & 15;
  const int hlf  = lane >> 4;

  const int e  = blockIdx.x >> 7;           // 128 token-tiles per expert
  const int t0 = (blockIdx.x & 127) * TILE_T;

  const float* xe  = x  + ((size_t)e * T_DIM + t0) * D_DIM;
  const float* w1e = w1 + (size_t)e * D_DIM * H_DIM;
  const float* b1e = b1 + (size_t)e * H_DIM;
  const float* w2e = w2 + (size_t)e * H_DIM * D_DIM;
  const float* b2e = b2 + (size_t)e * D_DIM;
  float*      oute = out + ((size_t)e * T_DIM + t0) * D_DIM;

  stage_x(xe, x_lds, tid);

  // ============ Phase 1: act = relu(x @ W1 + b1), kept bf16 in LDS ============
#pragma unroll 1
  for (int hc = 0; hc < H_DIM / CHUNK; ++hc) {
    const int c0 = hc * CHUNK;
    v8f acc[NT];
#pragma unroll
    for (int i = 0; i < NT; ++i) acc[i] = (v8f){0, 0, 0, 0, 0, 0, 0, 0};

    {  // prologue: stage slab ks=0 into buffer 0
      Staged s;
      slab_load(s, w1e + c0, H_DIM, tid);
      slab_store(s, slab0, tid);
    }
    __syncthreads();  // slab0 (and x_lds on hc==0) visible

#pragma unroll 2     // makes cur/nxt compile-time, one barrier per iter
    for (int ks = 0; ks < D_DIM / KSTEP; ++ks) {
      __bf16* cur = (ks & 1) ? slab1 : slab0;
      __bf16* nxt = (ks & 1) ? slab0 : slab1;
      const bool more = (ks + 1) < (D_DIM / KSTEP);
      Staged s;
      if (more)  // issue next slab's global loads before compute
        slab_load(s, w1e + (size_t)((ks + 1) * KSTEP) * H_DIM + c0, H_DIM, tid);

      const v16bf a = load_frag_a(x_lds, XLDS_STRIDE, ks * KSTEP, lane);
#pragma unroll
      for (int nt = 0; nt < NT; ++nt) {
        const v16bf bb = load_frag_b(cur, wave * 64 + nt * 16, lane);
        acc[nt] = __builtin_amdgcn_wmma_f32_16x16x32_bf16(
            false, a, false, bb, (short)0, acc[nt], false, false);
      }

      if (more) slab_store(s, nxt, tid);
      __syncthreads();  // cur fully read by all waves; nxt visible
    }

    // Epilogue: bias + ReLU, write bf16 activation tile to LDS.
#pragma unroll
    for (int nt = 0; nt < NT; ++nt) {
      const int   hb   = c0 + wave * 64 + nt * 16;
      const float bias = b1e[hb + n];
#pragma unroll
      for (int v = 0; v < 8; ++v) {
        float r = acc[nt][v] + bias;
        r = fmaxf(r, 0.0f);
        act_lds[(size_t)(8 * hlf + v) * ACT_STRIDE + hb + n] = (__bf16)r;
      }
    }
  }

  // ============ Phase 2: out = act @ W2 + b2 ============
#pragma unroll 1
  for (int dc = 0; dc < D_DIM / CHUNK; ++dc) {
    const int c0 = dc * CHUNK;
    v8f acc[NT];
#pragma unroll
    for (int i = 0; i < NT; ++i) acc[i] = (v8f){0, 0, 0, 0, 0, 0, 0, 0};

    {  // prologue (its barrier also orders phase-1 act writes before reads)
      Staged s;
      slab_load(s, w2e + c0, D_DIM, tid);
      slab_store(s, slab0, tid);
    }
    __syncthreads();

#pragma unroll 2
    for (int ks = 0; ks < H_DIM / KSTEP; ++ks) {
      __bf16* cur = (ks & 1) ? slab1 : slab0;
      __bf16* nxt = (ks & 1) ? slab0 : slab1;
      const bool more = (ks + 1) < (H_DIM / KSTEP);
      Staged s;
      if (more)
        slab_load(s, w2e + (size_t)((ks + 1) * KSTEP) * D_DIM + c0, D_DIM, tid);

      const v16bf a = load_frag_a(act_lds, ACT_STRIDE, ks * KSTEP, lane);
#pragma unroll
      for (int nt = 0; nt < NT; ++nt) {
        const v16bf bb = load_frag_b(cur, wave * 64 + nt * 16, lane);
        acc[nt] = __builtin_amdgcn_wmma_f32_16x16x32_bf16(
            false, a, false, bb, (short)0, acc[nt], false, false);
      }

      if (more) slab_store(s, nxt, tid);
      __syncthreads();
    }

    // Epilogue: bias + fp32 store (coalesced 64B per half-wave).
#pragma unroll
    for (int nt = 0; nt < NT; ++nt) {
      const int   db   = c0 + wave * 64 + nt * 16;
      const float bias = b2e[db + n];
#pragma unroll
      for (int v = 0; v < 8; ++v) {
        oute[(size_t)(8 * hlf + v) * D_DIM + db + n] = acc[nt][v] + bias;
      }
    }
  }
}

extern "C" void kernel_launch(void* const* d_in, const int* in_sizes, int n_in,
                              void* d_out, int out_size, void* d_ws, size_t ws_size,
                              hipStream_t stream) {
  (void)in_sizes; (void)n_in; (void)out_size; (void)d_ws; (void)ws_size;
  const float* x   = (const float*)d_in[0];
  const float* w1  = (const float*)d_in[1];
  const float* b1  = (const float*)d_in[2];
  const float* w2  = (const float*)d_in[3];
  const float* b2  = (const float*)d_in[4];
  float*       out = (float*)d_out;

  const dim3 grid(E_DIM * (T_DIM / TILE_T));  // 1024 workgroups
  moe_ffn_fused_kernel<<<grid, dim3(THREADS), LDS_TOTAL, stream>>>(
      x, w1, b1, w2, b2, out);
}
